// E75MultiHeadCell_20547123544802
// MI455X (gfx1250) — matrix operand order repaired and checked by reading, not compiled, not run on hardware
//
#include <hip/hip_runtime.h>
#include <hip/hip_bf16.h>

typedef __bf16 bf16_t;
typedef __attribute__((ext_vector_type(4)))  __bf16 v4bf;
typedef __attribute__((ext_vector_type(8)))  __bf16 v8bf;
typedef __attribute__((ext_vector_type(16))) __bf16 v16bf;
typedef __attribute__((ext_vector_type(8)))  float  v8f;

static constexpr int T_   = 1024;
static constexpr int B_   = 16;
static constexpr int DIM_ = 1024;
static constexpr int H_   = 8;
static constexpr int N_   = 64;
static constexpr int M_   = T_ * B_;    // 16384 rows
static constexpr int HN_  = H_ * N_;    // 512 cols
static constexpr int MT   = M_ / 16;    // 1024 m-tiles
static constexpr int NT   = HN_ / 16;   // 32 n-tiles

__device__ __forceinline__ bf16_t f2bf(float f) {
    unsigned u = __builtin_bit_cast(unsigned, f);
    unsigned short r = (unsigned short)((u + 0x7FFFu + ((u >> 16) & 1u)) >> 16); // RNE
    return __builtin_bit_cast(bf16_t, r);
}

// tanh via hardware exp; sign-safe (no inf/inf on negative inputs)
__device__ __forceinline__ float fast_tanh(float x) {
    float ax = __builtin_fabsf(x);
    float e  = __expf(-2.0f * ax);
    float t  = (1.0f - e) / (1.0f + e);
    return __builtin_copysignf(t, x);
}

// ---------------------------------------------------------------------------
// Phase 0: one-time fp32 -> bf16 conversion passes (memory-bound, ~5us).
// Removes all conversion VALU work from the GEMM inner loop.
// ---------------------------------------------------------------------------
__global__ __launch_bounds__(256) void cvt_x(const float* __restrict__ src,
                                             bf16_t* __restrict__ dst)
{
    const size_t idx = (size_t)blockIdx.x * 256 + threadIdx.x;   // float4 index
    const float4 v = ((const float4*)src)[idx];
    v4bf o;
    o[0] = f2bf(v.x); o[1] = f2bf(v.y); o[2] = f2bf(v.z); o[3] = f2bf(v.w);
    ((v4bf*)dst)[idx] = o;
}

__global__ __launch_bounds__(256) void cvt_w(
    const float* __restrict__ Wk, const float* __restrict__ Wv,
    const float* __restrict__ Wq, const float* __restrict__ Wb,
    bf16_t* __restrict__ dst)
{
    const int widx = blockIdx.y;
    const float* src = (widx == 0) ? Wk : (widx == 1) ? Wv : (widx == 2) ? Wq : Wb;
    const size_t idx = (size_t)blockIdx.x * 256 + threadIdx.x;   // float4 index
    const float4 v = ((const float4*)src)[idx];
    v4bf o;
    o[0] = f2bf(v.x); o[1] = f2bf(v.y); o[2] = f2bf(v.z); o[3] = f2bf(v.w);
    ((v4bf*)(dst + (size_t)widx * HN_ * DIM_))[idx] = o;
}

// ---------------------------------------------------------------------------
// Phase 1: fused projection GEMMs on pre-converted bf16 operands.
// One wave computes a 16x16 fp32 tile of one weight's output; K=1024 in chunks
// of 32 via v_wmma_f32_16x16x32_bf16.  Fragment gathers are contiguous 16-byte
// global_load_b128s per lane (bf16 A/B wave layouts are 8-element runs).
// A/B working set (37.5 MB bf16) lives in the 192 MB L2.  q's output goes
// straight into d_out; beta fuses bias+sigmoid.
// ---------------------------------------------------------------------------
__global__ __launch_bounds__(256) void proj_gemm(
    const bf16_t* __restrict__ xbf, const bf16_t* __restrict__ wbf,
    const float* __restrict__ bbeta,
    float* __restrict__ ok, float* __restrict__ ov,
    float* __restrict__ oq, float* __restrict__ ob)
{
    const int wave  = (blockIdx.x << 3) | (threadIdx.x >> 5);
    const int lid   = threadIdx.x & 31;
    // decompose so the 8 waves of a block share the same m-tile (A locality)
    const int mtile = wave >> 7;          // 0..1023
    const int rem   = wave & 127;
    const int widx  = rem >> 5;           // 0..3  (k, v, q, beta)
    const int ntile = rem & 31;           // 0..31

    float* dst = (widx == 0) ? ok : (widx == 1) ? ov : (widx == 2) ? oq : ob;

    const int mrow = (mtile << 4) + (lid & 15);
    const int ncol = (ntile << 4) + (lid & 15);
    const int half = lid >> 4;            // 0: lanes 0-15, 1: lanes 16-31

    const bf16_t* xrow = xbf + (size_t)mrow * DIM_;
    const bf16_t* wrow = wbf + ((size_t)widx * HN_ + ncol) * DIM_;

    v8f c = {};
    #pragma unroll 4
    for (int kb = 0; kb < DIM_; kb += 32) {
        // A 16x32 bf16 layout: elems 0..7  -> K = kb + half*8 + e
        //                      elems 8..15 -> K = kb + 16 + half*8 + e
        v8bf a0 = *(const v8bf*)(xrow + kb + half * 8);
        v8bf a1 = *(const v8bf*)(xrow + kb + 16 + half * 8);
        // B 32x16 bf16 layout: same K mapping as A, N = lid&15
        v8bf b0 = *(const v8bf*)(wrow + kb + half * 8);
        v8bf b1 = *(const v8bf*)(wrow + kb + 16 + half * 8);

        v16bf a = __builtin_shufflevector(a0, a1, 0,1,2,3,4,5,6,7,
                                                  8,9,10,11,12,13,14,15);
        v16bf b = __builtin_shufflevector(b0, b1, 0,1,2,3,4,5,6,7,
                                                  8,9,10,11,12,13,14,15);
        c = __builtin_amdgcn_wmma_f32_16x16x32_bf16(false, a, false, b,
                                                    (short)0, c, false, false);
    }

    // C layout: VGPR r -> row (mtile*16 + half*8 + r), col = ncol
    const int crow = (mtile << 4) + (half << 3);
    if (widx == 3) {
        const float bias = bbeta[ncol];
        #pragma unroll
        for (int r = 0; r < 8; ++r) {
            float v = c[r] + bias;
            v = 1.0f / (1.0f + __expf(-v));
            dst[(size_t)(crow + r) * HN_ + ncol] = v;
        }
    } else {
        #pragma unroll
        for (int r = 0; r < 8; ++r)
            dst[(size_t)(crow + r) * HN_ + ncol] = c[r];
    }
}

// ---------------------------------------------------------------------------
// Phase 2: persistent sequential scan.  One block per (b,h) cell; 64x64 state S
// lives in registers (thread t owns row i=t/4, 16-col slice j0=(t&3)*16).
// Per step: stage k/v/q/beta (4x64 floats) to LDS, row reductions over the
// 4 owning lanes via wave32 shfl_xor.  q is read from d_out and overwritten
// with the step output (read happens before the write within each step).
// ---------------------------------------------------------------------------
__global__ __launch_bounds__(256) void delta_scan(
    const float* __restrict__ S0,
    const float* __restrict__ kall, const float* __restrict__ vall,
    const float* __restrict__ qall, const float* __restrict__ ball,
    float* __restrict__ out, float* __restrict__ Sfin)
{
    const int bh  = blockIdx.x;          // b*8 + h
    const int b   = bh >> 3;
    const int h   = bh & 7;
    const int tid = threadIdx.x;
    const int i   = tid >> 2;            // state row 0..63
    const int c4  = tid & 3;             // column-slice id
    const int j0  = c4 << 4;             // first column of slice

    float S[16];
    {
        const float* S0p = S0 + (size_t)bh * (N_ * N_) + i * N_ + j0;
        #pragma unroll
        for (int e = 0; e < 16; ++e) S[e] = S0p[e];
    }

    __shared__ float sk[64], sv[64], sq[64], sb[64];

    const int lane64 = tid & 63;
    const int which  = tid >> 6;
    const float* mysrc = (which == 0) ? kall : (which == 1) ? vall
                        : (which == 2) ? qall : ball;
    float* mydst = (which == 0) ? sk : (which == 1) ? sv
                  : (which == 2) ? sq : sb;

    size_t off = (size_t)b * HN_ + (size_t)h * N_;
    const size_t stride = (size_t)B_ * HN_;          // 8192 floats per step

    for (int t = 0; t < T_; ++t, off += stride) {
        mydst[lane64] = mysrc[off + lane64];
        __syncthreads();

        // k_norm = k / (||k|| + 1e-6)
        float kn[16];
        float p = 0.f;
        #pragma unroll
        for (int e = 0; e < 16; ++e) { float kv = sk[j0 + e]; kn[e] = kv; p += kv * kv; }
        p += __shfl_xor(p, 1, 32);
        p += __shfl_xor(p, 2, 32);
        const float inv = 1.0f / (__builtin_sqrtf(p) + 1e-6f);

        // retrieved_i = S[i,:] . k_norm
        float r = 0.f;
        #pragma unroll
        for (int e = 0; e < 16; ++e) { kn[e] *= inv; r += S[e] * kn[e]; }
        r += __shfl_xor(r, 1, 32);
        r += __shfl_xor(r, 2, 32);

        const float delta = sv[i] - r;
        const float beta  = sb[i];

        // S = tanh(beta*S + delta (x) k_norm);  Sq_i = S[i,:] . q
        float sqp = 0.f;
        #pragma unroll
        for (int e = 0; e < 16; ++e) {
            float s = fast_tanh(beta * S[e] + delta * kn[e]);
            S[e] = s;
            sqp += s * sq[j0 + e];
        }
        sqp += __shfl_xor(sqp, 1, 32);
        sqp += __shfl_xor(sqp, 2, 32);

        if (c4 == 0) {
            const float sg = 1.0f / (1.0f + __expf(-sqp));
            out[off + i] = sqp * sqp * sg;           // Sq * silu(Sq)
        }
        __syncthreads();
    }

    float* Sf = Sfin + (size_t)bh * (N_ * N_) + i * N_ + j0;
    #pragma unroll
    for (int e = 0; e < 16; ++e) Sf[e] = S[e];
}

extern "C" void kernel_launch(void* const* d_in, const int* in_sizes, int n_in,
                              void* d_out, int out_size, void* d_ws, size_t ws_size,
                              hipStream_t stream) {
    const float* x  = (const float*)d_in[0];
    const float* S0 = (const float*)d_in[1];
    const float* Wk = (const float*)d_in[2];
    const float* Wv = (const float*)d_in[3];
    const float* Wq = (const float*)d_in[4];
    const float* Wb = (const float*)d_in[5];
    const float* bb = (const float*)d_in[6];

    float* out  = (float*)d_out;                       // [T,B,HN]
    float* Sfin = out + (size_t)T_ * B_ * HN_;         // [B,H,N,N]

    // workspace layout (floats): k | v | beta | xbf | wbf
    float* ws = (float*)d_ws;
    const size_t per = (size_t)M_ * HN_;               // 8M floats each
    float* ok = ws;
    float* ov = ws + per;
    float* ob = ws + 2 * per;
    float* oq = out;                                   // q staged in d_out, scan overwrites
    bf16_t* xbf = (bf16_t*)(ws + 3 * per);                      // 16384x1024 bf16
    bf16_t* wbf = (bf16_t*)((char*)xbf + (size_t)M_ * DIM_ * sizeof(bf16_t)); // 4x512x1024

    cvt_x<<<(M_ * DIM_) / 4 / 256, 256, 0, stream>>>(x, xbf);
    cvt_w<<<dim3((HN_ * DIM_) / 4 / 256, 4), 256, 0, stream>>>(Wk, Wv, Wq, Wb, wbf);

    const int gemm_blocks = (MT * NT * 4) / 8;         // 16384 blocks x 8 waves
    proj_gemm<<<gemm_blocks, 256, 0, stream>>>(xbf, wbf, bb, ok, ov, oq, ob);
    delta_scan<<<B_ * H_, 256, 0, stream>>>(S0, ok, ov, oq, ob, out, Sfin);
}
